// ICL_67448166416614
// MI455X (gfx1250) — matrix operand order repaired and checked by reading, not compiled, not run on hardware
//
#include <hip/hip_runtime.h>
#include <math.h>

// ---------------------------------------------------------------------------
// MI455X (gfx1250, wave32) implementation.
//
// Pipeline (all bf16 WMMA 16x16x32, f32 accumulate):
//   For each of the 6 MLP layers:
//     gemm_layer: A = act(bn(h_prev)) (fused on load, or windowed gather of x
//                 for layer 0), B = W (staged [N][K] in LDS), D -> pre-BN bf16.
//     rowstats:   per-row sum / sumsq (deterministic, no atomics)
//     finstats:   per-window mean / rstd (151 scalars)
//   final_kernel (1 block per batch element):
//     bn+leaky both 151x128 matrices into LDS, per-column (d) L2 norms over
//     the 151 windows folded into the K dimension, S = Q * P^T via 10x10
//     WMMA tiles, scatter through inverse negative_index permutation with
//     diagonal -> out[b,0,n] and -inf masking; output already transposed.
// ---------------------------------------------------------------------------

#define NWIN  151
#define NBATCH 2048
#define M_TOT 309248          // NBATCH * NWIN
#define NFEAT 200
#define KERN  50

typedef __attribute__((ext_vector_type(16))) __bf16 v16bf;
typedef __attribute__((ext_vector_type(8)))  __bf16 v8bf;
typedef __attribute__((ext_vector_type(8)))  float  v8f;

__device__ __forceinline__ v8f wmma_bf16(v16bf a, v16bf b, v8f c) {
  // v_wmma_f32_16x16x32_bf16
  return __builtin_amdgcn_wmma_f32_16x16x32_bf16(
      /*neg_a=*/false, a, /*neg_b=*/false, b,
      /*c_mod=*/(short)0, c, /*reuse_a=*/false, /*reuse_b=*/false);
}

// A operand (16x32, MxK) from row-major [M][ldk] LDS tile (base already at +ks).
// Lane layout (ISA 7.12.2): lanes 0-15 row=lane, elems 0..7 -> K=0..7,
// elems 8..15 -> K=16..23 ; lanes 16-31 shifted by +8.
__device__ __forceinline__ v16bf ld_frag_mk(const __bf16* base, int ldk) {
  int lane = threadIdx.x & 31;
  int row  = lane & 15;
  int hi   = (lane >> 4) & 1;
  const __bf16* p = base + row * ldk + (hi << 3);
  union { v16bf v; v8bf h[2]; } u;
  u.h[0] = *(const v8bf*)(p);
  u.h[1] = *(const v8bf*)(p + 16);
  return u.v;
}

// B operand (32x16, KxN) from row-major [N][ldk] LDS tile (i.e. W as (out,in)).
// Lanes 0-15: N=lane, elems 0..7 -> K=0..7, elems 8..15 -> K=16..23 ; lanes 16-31 shifted by +8.
__device__ __forceinline__ v16bf ld_frag_nk(const __bf16* base, int ldk) {
  int lane = threadIdx.x & 31;
  int col  = lane & 15;
  int hi   = (lane >> 4) & 1;
  const __bf16* p = base + col * ldk + (hi << 3);
  union { v16bf v; v8bf h[2]; } u;
  u.h[0] = *(const v8bf*)(p);
  u.h[1] = *(const v8bf*)(p + 16);
  return u.v;
}

// ---------------------------------------------------------------------------
// Layer GEMM: D[M_TOT x Nreal] = A[M_TOT x Kreal] * W^T, pre-BN output (bf16).
// GATHER: 0 = from hprev with fused bn+act, 1 = x1 window gather, 2 = x2
// complement gather. ACT (prev layer's activation): 0 none, 1 leaky, 2 tanh.
// Block = 128 threads = 4 waves; each wave owns 16 rows; W shared in LDS.
// ---------------------------------------------------------------------------
template <int KP, int NT, int ACT, int GATHER>
__global__ void __launch_bounds__(128)
gemm_layer(const float* __restrict__ x, const __bf16* __restrict__ hprev,
           const float* __restrict__ stats, const float* __restrict__ Wg,
           __bf16* __restrict__ hout, int Kreal, int Nreal) {
  extern __shared__ __bf16 lds[];
  __bf16* Wl = lds;                       // [NT*16][KP]
  __bf16* Al = lds + NT * 16 * KP;        // [4][16][KP]
  const int tid  = threadIdx.x;
  const int lane = tid & 31;
  const int wave = tid >> 5;

  // Stage weights (zero-padded to tile multiples).
  for (int i = tid; i < NT * 16 * KP; i += 128) {
    int r = i / KP, k = i - r * KP;
    float v = (r < Nreal && k < Kreal) ? Wg[r * Kreal + k] : 0.0f;
    Wl[i] = (__bf16)v;
  }

  // Stage this wave's 16-row A tile with fused gather / bn+activation.
  __bf16* Aw = Al + wave * 16 * KP;
  const int rowBase = blockIdx.x * 64 + wave * 16;   // M_TOT % 64 == 0
  for (int i = 0; i < 16; ++i) {
    int row = rowBase + i;
    int n   = row % NWIN;
    for (int k = lane; k < KP; k += 32) {
      float v = 0.0f;
      if (k < Kreal) {
        if (GATHER == 1) {                       // x1: window [n, n+49]
          int b = row / NWIN;
          v = x[b * NFEAT + n + k];
        } else if (GATHER == 2) {                // x2: complement of window
          int b = row / NWIN;
          v = x[b * NFEAT + (k < n ? k : k + KERN)];
        } else {                                 // bn + act of previous layer
          v = (float)hprev[(size_t)row * Kreal + k];
          v = (v - stats[n * 2]) * stats[n * 2 + 1];
          if (ACT == 1) v = v >= 0.0f ? v : 0.01f * v;
          else if (ACT == 2) v = tanhf(v);
        }
      }
      Aw[i * KP + k] = (__bf16)v;
    }
  }
  __syncthreads();

  // 16x16 output tiles across N; K accumulated in steps of 32.
  for (int nt = 0; nt < NT; ++nt) {
    v8f acc = {0, 0, 0, 0, 0, 0, 0, 0};
    const __bf16* Bp = Wl + nt * 16 * KP;
#pragma unroll
    for (int ks = 0; ks < KP; ks += 32) {
      v16bf a = ld_frag_mk(Aw + ks, KP);
      v16bf b = ld_frag_nk(Bp + ks, KP);
      acc = wmma_bf16(a, b, acc);
    }
    int c = nt * 16 + (lane & 15);
    if (c < Nreal) {
      int rb = rowBase + ((lane >> 4) << 3);     // lanes16-31 hold M = v + 8
#pragma unroll
      for (int v = 0; v < 8; ++v)
        hout[(size_t)(rb + v) * Nreal + c] = (__bf16)acc[v];
    }
  }
}

// Per-row sum / sumsq (deterministic: one thread per row, fixed order).
__global__ void __launch_bounds__(256)
rowstats(const __bf16* __restrict__ h, float* __restrict__ rs, int Nf) {
  int row = blockIdx.x * blockDim.x + threadIdx.x;
  if (row >= M_TOT) return;
  const __bf16* p = h + (size_t)row * Nf;
  float s = 0.0f, s2 = 0.0f;
  for (int k = 0; k < Nf; ++k) {
    float v = (float)p[k];
    s += v;
    s2 += v * v;
  }
  rs[row * 2]     = s;
  rs[row * 2 + 1] = s2;
}

// Reduce row partials -> per-window (mean, rstd). BN over axes (0,2).
__global__ void __launch_bounds__(160)
finstats(const float* __restrict__ rs, float* __restrict__ st, float invCount) {
  int n = threadIdx.x;
  if (n >= NWIN) return;
  float s = 0.0f, s2 = 0.0f;
  for (int b = 0; b < NBATCH; ++b) {
    s  += rs[(b * NWIN + n) * 2];
    s2 += rs[(b * NWIN + n) * 2 + 1];
  }
  float m   = s * invCount;
  float var = s2 * invCount - m * m;
  st[n * 2]     = m;
  st[n * 2 + 1] = rsqrtf(var + 1e-5f);
}

// ---------------------------------------------------------------------------
// Final per-batch kernel: bn+leaky, window-axis normalization (folded into K),
// S = Q * P^T via WMMA, permuted scatter with diagonal masking.
// Dynamic LDS = 2*(160*128 bf16) + 256 f32 + 160 int = 83584 B (<320KB/WGP).
// ---------------------------------------------------------------------------
__global__ void __launch_bounds__(256)
final_kernel(const __bf16* __restrict__ hg2, const __bf16* __restrict__ hf2,
             const float* __restrict__ sg2, const float* __restrict__ sf2,
             const int* __restrict__ negidx, float* __restrict__ out) {
  extern __shared__ char sm[];
  __bf16* Q  = (__bf16*)sm;                 // query tile [160][128]
  __bf16* P  = Q + 160 * 128;               // pos tile   [160][128]
  float*  cn = (float*)(P + 160 * 128);     // [0..127]=1/||Q_d||, [128..255]=P
  int*    inv = (int*)(cn + 256);           // inverse permutation

  const int b   = blockIdx.x;
  const int tid = threadIdx.x;
  const __bf16* qg = hf2 + (size_t)b * NWIN * 128;
  const __bf16* pg = hg2 + (size_t)b * NWIN * 128;

  for (int j = tid; j < NWIN; j += 256) inv[negidx[j]] = j;

  for (int idx = tid; idx < 160 * 128; idx += 256) {
    int r = idx >> 7, c = idx & 127;
    float vq = 0.0f, vp = 0.0f;
    if (r < NWIN) {
      float v = (float)qg[r * 128 + c];
      v  = (v - sf2[r * 2]) * sf2[r * 2 + 1];
      vq = v >= 0.0f ? v : 0.01f * v;
      v  = (float)pg[r * 128 + c];
      v  = (v - sg2[r * 2]) * sg2[r * 2 + 1];
      vp = v >= 0.0f ? v : 0.01f * v;
    }
    Q[idx] = (__bf16)vq;
    P[idx] = (__bf16)vp;
  }
  __syncthreads();

  // Deterministic per-column (rep-dim) norms over the 151 windows.
  if (tid < 256) {
    const __bf16* base = (tid < 128) ? Q : P;
    int c = tid & 127;
    float s = 0.0f;
    for (int r = 0; r < NWIN; ++r) {
      float v = (float)base[r * 128 + c];
      s += v * v;
    }
    cn[tid] = 1.0f / fmaxf(sqrtf(s), 1e-12f);
  }
  __syncthreads();
  for (int idx = tid; idx < 160 * 128; idx += 256) {
    int c = idx & 127;
    Q[idx] = (__bf16)((float)Q[idx] * cn[c]);
    P[idx] = (__bf16)((float)P[idx] * cn[128 + c]);
  }
  __syncthreads();

  // S = Q * P^T : 10x10 grid of 16x16 tiles, K = 128 (4 wmma each).
  const int wave = tid >> 5;
  const int lane = tid & 31;
  float* ob = out + (size_t)b * 152 * NWIN;
  for (int t = wave; t < 100; t += 8) {
    int i0 = (t / 10) * 16, j0 = (t % 10) * 16;
    v8f acc = {0, 0, 0, 0, 0, 0, 0, 0};
#pragma unroll
    for (int ks = 0; ks < 128; ks += 32) {
      v16bf a  = ld_frag_mk(Q + i0 * 128 + ks, 128);
      v16bf bb = ld_frag_nk(P + j0 * 128 + ks, 128);
      acc = wmma_bf16(a, bb, acc);
    }
    int m = j0 + (lane & 15);                 // pos window index
    if (m < NWIN) {
      int im = inv[m];
      int rb = i0 + ((lane >> 4) << 3);
#pragma unroll
      for (int v = 0; v < 8; ++v) {
        int n = rb + v;                        // query window index
        if (n < NWIN) {
          float val = acc[v] * 100.0f;         // / TAU
          if (m == n) {
            ob[n] = val;                                        // out[b,0,n]
            ob[(size_t)(1 + im) * NWIN + n] = -__builtin_inff();// masked diag
          } else {
            ob[(size_t)(1 + im) * NWIN + n] = val;
          }
        }
      }
    }
  }
}

// ---------------------------------------------------------------------------
extern "C" void kernel_launch(void* const* d_in, const int* in_sizes, int n_in,
                              void* d_out, int out_size, void* d_ws,
                              size_t ws_size, hipStream_t stream) {
  (void)in_sizes; (void)n_in; (void)out_size; (void)ws_size;
  const float* x   = (const float*)d_in[0];
  const float* gW0 = (const float*)d_in[1];
  const float* gW1 = (const float*)d_in[2];
  const float* gW2 = (const float*)d_in[3];
  const float* fW0 = (const float*)d_in[4];
  const float* fW1 = (const float*)d_in[5];
  const float* fW2 = (const float*)d_in[6];
  const int*   ni  = (const int*)d_in[7];
  float* out = (float*)d_out;

  // Workspace carving (~300 MB; all regions fully rewritten every call).
  char* base = (char*)d_ws;
  size_t off = 0;
  auto take = [&](size_t bytes) -> char* {
    char* p = base + off;
    off += (bytes + 255) & ~(size_t)255;
    return p;
  };
  float*  rowst = (float*)take((size_t)M_TOT * 2 * 4);
  float*  stats = (float*)take((size_t)6 * 302 * 4);   // 6 layers x (mean,rstd)
  __bf16* hg0 = (__bf16*)take((size_t)M_TOT * 50  * 2);
  __bf16* hg1 = (__bf16*)take((size_t)M_TOT * 25  * 2);
  __bf16* hg2 = (__bf16*)take((size_t)M_TOT * 128 * 2);
  __bf16* hf0 = (__bf16*)take((size_t)M_TOT * 100 * 2);
  __bf16* hf1 = (__bf16*)take((size_t)M_TOT * 50  * 2);
  __bf16* hf2 = (__bf16*)take((size_t)M_TOT * 128 * 2);

  const int GB = M_TOT / 64;       // 4832 GEMM blocks
  const int RB = M_TOT / 256;      // 1208 rowstats blocks

  // ---- g path: 50 -> 50 -> 25 -> 128 (leaky throughout) ----
  gemm_layer<64, 4, 0, 1><<<GB, 128, (4 * 16 * 64 + 64 * 64) * 2, stream>>>(
      x, nullptr, nullptr, gW0, hg0, 50, 50);
  rowstats<<<RB, 256, 0, stream>>>(hg0, rowst, 50);
  finstats<<<1, 160, 0, stream>>>(rowst, stats + 0 * 302, 1.0f / (2048.0f * 50.0f));

  gemm_layer<64, 2, 1, 0><<<GB, 128, (2 * 16 * 64 + 64 * 64) * 2, stream>>>(
      nullptr, hg0, stats + 0 * 302, gW1, hg1, 50, 25);
  rowstats<<<RB, 256, 0, stream>>>(hg1, rowst, 25);
  finstats<<<1, 160, 0, stream>>>(rowst, stats + 1 * 302, 1.0f / (2048.0f * 25.0f));

  gemm_layer<32, 8, 1, 0><<<GB, 128, (8 * 16 * 32 + 64 * 32) * 2, stream>>>(
      nullptr, hg1, stats + 1 * 302, gW2, hg2, 25, 128);
  rowstats<<<RB, 256, 0, stream>>>(hg2, rowst, 128);
  finstats<<<1, 160, 0, stream>>>(rowst, stats + 2 * 302, 1.0f / (2048.0f * 128.0f));

  // ---- f path: 150 -> 100 -> 50 -> 128 (tanh, leaky, leaky) ----
  gemm_layer<160, 7, 0, 2><<<GB, 128, (7 * 16 * 160 + 64 * 160) * 2, stream>>>(
      x, nullptr, nullptr, fW0, hf0, 150, 100);
  rowstats<<<RB, 256, 0, stream>>>(hf0, rowst, 100);
  finstats<<<1, 160, 0, stream>>>(rowst, stats + 3 * 302, 1.0f / (2048.0f * 100.0f));

  gemm_layer<128, 4, 2, 0><<<GB, 128, (4 * 16 * 128 + 64 * 128) * 2, stream>>>(
      nullptr, hf0, stats + 3 * 302, fW1, hf1, 100, 50);
  rowstats<<<RB, 256, 0, stream>>>(hf1, rowst, 50);
  finstats<<<1, 160, 0, stream>>>(rowst, stats + 4 * 302, 1.0f / (2048.0f * 50.0f));

  gemm_layer<64, 8, 1, 0><<<GB, 128, (8 * 16 * 64 + 64 * 64) * 2, stream>>>(
      nullptr, hf1, stats + 4 * 302, fW2, hf2, 50, 128);
  rowstats<<<RB, 256, 0, stream>>>(hf2, rowst, 128);
  finstats<<<1, 160, 0, stream>>>(rowst, stats + 5 * 302, 1.0f / (2048.0f * 128.0f));

  // ---- final similarity + permuted scatter ----
  const size_t FINAL_LDS = (size_t)160 * 128 * 2 * 2 + 256 * 4 + 160 * 4;
  final_kernel<<<NBATCH, 256, FINAL_LDS, stream>>>(hg2, hf2, stats + 2 * 302,
                                                   stats + 5 * 302, ni, out);
}
